// SelfAttention_59296318488697
// MI455X (gfx1250) — matrix operand order repaired and checked by reading, not compiled, not run on hardware
//
#include <hip/hip_runtime.h>
#include <math.h>

typedef __bf16 bf16_t;
typedef __attribute__((ext_vector_type(16))) __bf16 v16bf;
typedef __attribute__((ext_vector_type(8)))  __bf16 v8bf;
typedef __attribute__((ext_vector_type(8)))  float  v8f;
typedef __attribute__((ext_vector_type(4))) unsigned int u32x4;
typedef __attribute__((ext_vector_type(8))) int i32x8;
typedef __attribute__((ext_vector_type(4))) int i32x4;

#define B_    2
#define S_    2048
#define HID_  2048
#define H_    16
#define KVH_  4
#define NOPE_ 128
#define ROPE_ 64
#define VD_   128
#define QR_   1536
#define KVR_  512
#define QD_   (NOPE_ + ROPE_)          // 192
#define KVD_  (NOPE_ + ROPE_ + VD_)    // 320
#define QOUT_ (H_ * QD_)               // 3072
#define KVOUT_ (KVH_ * KVD_)           // 1280
#define BS_   (B_ * S_)                // 4096

#define HAVE_TDM __has_builtin(__builtin_amdgcn_tensor_load_to_lds)

// ---------------------------------------------------------------------------
// Tensor Data Mover: 2-D tile load (global -> LDS), bf16 elements.
// D# packing per CDNA5 ISA ch.8: group0 = {count/lds_addr/global_addr/type},
// group1 = {data_size, tensor dims, tile dims, dim0 stride}. 2-D => groups 2/3
// zero. Issued by one wave; completion via TENSORcnt.
// ---------------------------------------------------------------------------
#if HAVE_TDM
static __device__ __forceinline__ void tdm_load_2d(
    unsigned lds_addr, unsigned long long gaddr,
    unsigned tensor_d0, unsigned tensor_d1,
    unsigned long long stride0, unsigned tile_d0, unsigned tile_d1) {
  u32x4 g0;
  g0[0] = 1u;                                         // count=1, no gather
  g0[1] = lds_addr;                                   // LDS byte address
  g0[2] = (unsigned)(gaddr & 0xffffffffull);          // global_addr[31:0]
  g0[3] = (unsigned)((gaddr >> 32) & 0x01ffffffull)   // global_addr[56:32]
          | 0x80000000u;                              // type=2 ("image")
  i32x8 g1;
  g1[0] = 0x00010000;                                 // data_size=1 (2B), mask=0
  g1[1] = (int)((tensor_d0 & 0xffffu) << 16);         // tensor_dim0[15:0]
  g1[2] = (int)((tensor_d0 >> 16) | ((tensor_d1 & 0xffffu) << 16));
  g1[3] = (int)((tensor_d1 >> 16) | (tile_d0 << 16)); // tile_dim0
  g1[4] = (int)(tile_d1 & 0xffffu);                   // tile_dim1, tile_dim2=0
  g1[5] = (int)(stride0 & 0xffffffffull);             // dim0 stride[31:0]
  g1[6] = (int)((stride0 >> 32) & 0xffffull);         // stride[47:32], s1=0
  g1[7] = 0;
  i32x4 z4 = {0, 0, 0, 0};
#if defined(__clang_major__) && __clang_major__ >= 23
  i32x8 z8 = {0, 0, 0, 0, 0, 0, 0, 0};
  __builtin_amdgcn_tensor_load_to_lds(g0, g1, z4, z4, z8, 0);
#else
  __builtin_amdgcn_tensor_load_to_lds(g0, g1, z4, z4, 0);
#endif
}
#endif

// ---------------------------------------------------------------------------
// WMMA fragment helpers (wave32, 16x16x32 bf16 -> f32)
// ---------------------------------------------------------------------------
static __device__ __forceinline__ v16bf load_a_frag(const bf16_t* __restrict__ p,
                                                    int ld, int row0, int k0) {
  const int lane = threadIdx.x & 31;
  const bf16_t* q = p + (size_t)(row0 + (lane & 15)) * (size_t)ld
                      + (k0 + ((lane & 16) >> 1));
  union { v16bf v; v8bf h[2]; } u;
  u.h[0] = *(const v8bf*)(q);
  u.h[1] = *(const v8bf*)(q + 16);
  return u.v;
}

static __device__ __forceinline__ v16bf load_b_frag(const bf16_t* __restrict__ p,
                                                    int ld, int row0, int k0) {
  const int lane = threadIdx.x & 31;
  const bf16_t* q = p + (size_t)(row0 + (lane & 15)) * (size_t)ld
                      + (k0 + ((lane & 16) >> 1));
  union { v16bf v; v8bf h[2]; } u;
  u.h[0] = *(const v8bf*)(q);
  u.h[1] = *(const v8bf*)(q + 16);
  return u.v;
}

static __device__ __forceinline__ v8f wmma_bf16(v16bf a, v16bf b, v8f c) {
  return __builtin_amdgcn_wmma_f32_16x16x32_bf16(false, a, false, b,
                                                 (short)0, c, false, false);
}

// ---------------------------------------------------------------------------
// Direct-from-global bf16 GEMM (fallback path): C[M,N] f32 = A[M,K] * Bt[N,K]^T
// ---------------------------------------------------------------------------
__global__ __launch_bounds__(128) void gemm_bf16(const bf16_t* __restrict__ A,
                                                 const bf16_t* __restrict__ Bt,
                                                 float* __restrict__ C,
                                                 int M, int N, int K) {
  const int wave = threadIdx.x >> 5;
  const int lane = threadIdx.x & 31;
  const int m0 = blockIdx.y * 64 + ((wave >> 1) << 5);
  const int n0 = blockIdx.x * 64 + ((wave & 1) << 5);
  v8f acc[2][2] = {};
  for (int k = 0; k < K; k += 32) {
    if (k + 256 < K) {
      __builtin_prefetch(A  + (size_t)(m0 + lane) * K + k + 256, 0, 1);
      __builtin_prefetch(Bt + (size_t)(n0 + lane) * K + k + 256, 0, 1);
    }
    v16bf a0 = load_a_frag(A, K, m0,      k);
    v16bf a1 = load_a_frag(A, K, m0 + 16, k);
    v16bf b0 = load_b_frag(Bt, K, n0,      k);
    v16bf b1 = load_b_frag(Bt, K, n0 + 16, k);
    acc[0][0] = wmma_bf16(a0, b0, acc[0][0]);
    acc[0][1] = wmma_bf16(a0, b1, acc[0][1]);
    acc[1][0] = wmma_bf16(a1, b0, acc[1][0]);
    acc[1][1] = wmma_bf16(a1, b1, acc[1][1]);
  }
  const int col = lane & 15;
  const int rb  = (lane >> 4) << 3;
  for (int im = 0; im < 2; ++im)
    for (int jn = 0; jn < 2; ++jn)
      for (int i = 0; i < 8; ++i)
        C[(size_t)(m0 + im * 16 + rb + i) * (size_t)N + (n0 + jn * 16 + col)] =
            acc[im][jn][i];
}

#if HAVE_TDM
// ---------------------------------------------------------------------------
// TDM-staged bf16 GEMM: 64x64 block tile, K-step 64 staged in LDS by the TDM,
// consumed by 4 waves as ds_load_b128 fragments.
// ---------------------------------------------------------------------------
__global__ __launch_bounds__(128) void gemm_bf16_tdm(const bf16_t* __restrict__ A,
                                                     const bf16_t* __restrict__ Bt,
                                                     float* __restrict__ C,
                                                     int M, int N, int K) {
  __shared__ __align__(16) bf16_t As[64 * 64];   // 8 KB
  __shared__ __align__(16) bf16_t Bs[64 * 64];   // 8 KB
  const int wave = threadIdx.x >> 5;
  const int lane = threadIdx.x & 31;
  const int mBlk = blockIdx.y * 64;
  const int nBlk = blockIdx.x * 64;
  const int mo = (wave >> 1) << 5;
  const int no = (wave & 1) << 5;
  const unsigned ldsA = (unsigned)(unsigned long long)(uintptr_t)As;
  const unsigned ldsB = (unsigned)(unsigned long long)(uintptr_t)Bs;
  v8f acc[2][2] = {};
  for (int k = 0; k < K; k += 64) {
    if (wave == 0) {
      tdm_load_2d(ldsA,
                  (unsigned long long)(uintptr_t)(A + (size_t)mBlk * K + k),
                  (unsigned)(K - k), (unsigned)(M - mBlk),
                  (unsigned long long)K, 64u, 64u);
      tdm_load_2d(ldsB,
                  (unsigned long long)(uintptr_t)(Bt + (size_t)nBlk * K + k),
                  (unsigned)(K - k), (unsigned)(N - nBlk),
                  (unsigned long long)K, 64u, 64u);
      __builtin_amdgcn_s_wait_tensorcnt(0);
    }
    __syncthreads();
    for (int kk = 0; kk < 64; kk += 32) {
      v16bf a0 = load_a_frag(As, 64, mo,      kk);
      v16bf a1 = load_a_frag(As, 64, mo + 16, kk);
      v16bf b0 = load_b_frag(Bs, 64, no,      kk);
      v16bf b1 = load_b_frag(Bs, 64, no + 16, kk);
      acc[0][0] = wmma_bf16(a0, b0, acc[0][0]);
      acc[0][1] = wmma_bf16(a0, b1, acc[0][1]);
      acc[1][0] = wmma_bf16(a1, b0, acc[1][0]);
      acc[1][1] = wmma_bf16(a1, b1, acc[1][1]);
    }
    __syncthreads();
  }
  const int col = lane & 15;
  const int rb  = (lane >> 4) << 3;
  for (int im = 0; im < 2; ++im)
    for (int jn = 0; jn < 2; ++jn)
      for (int i = 0; i < 8; ++i)
        C[(size_t)(mBlk + mo + im * 16 + rb + i) * (size_t)N +
          (nBlk + no + jn * 16 + col)] = acc[im][jn][i];
}
#define GEMM_KERNEL gemm_bf16_tdm
#else
#define GEMM_KERNEL gemm_bf16
#endif

// ---------------------------------------------------------------------------
// Elementwise / layout kernels
// ---------------------------------------------------------------------------
__global__ void f32_to_bf16(const float* __restrict__ in, bf16_t* __restrict__ out,
                            size_t n) {
  size_t i = (size_t)blockIdx.x * blockDim.x + threadIdx.x;
  if (i < n) out[i] = (bf16_t)in[i];
}

// in: (R, Cn) row-major f32 -> out: (Cn, R) row-major bf16
__global__ void transpose_to_bf16(const float* __restrict__ in,
                                  bf16_t* __restrict__ out, int R, int Cn) {
  size_t idx = (size_t)blockIdx.x * blockDim.x + threadIdx.x;
  if (idx >= (size_t)R * Cn) return;
  int r = (int)(idx % R);
  size_t c = idx / R;
  out[idx] = (bf16_t)in[(size_t)r * Cn + c];
}

__global__ __launch_bounds__(256) void rmsnorm_bf16(const float* __restrict__ in,
                                                    const float* __restrict__ w,
                                                    bf16_t* __restrict__ out, int N) {
  __shared__ float red[256];
  const int row = blockIdx.x;
  const float* x = in + (size_t)row * N;
  float ss = 0.f;
  for (int i = threadIdx.x; i < N; i += 256) { float v = x[i]; ss += v * v; }
  red[threadIdx.x] = ss;
  __syncthreads();
  for (int s = 128; s > 0; s >>= 1) {
    if ((int)threadIdx.x < s) red[threadIdx.x] += red[threadIdx.x + s];
    __syncthreads();
  }
  const float scale = rsqrtf(red[0] / (float)N + 1e-6f);
  bf16_t* o = out + (size_t)row * N;
  for (int i = threadIdx.x; i < N; i += 256) o[i] = (bf16_t)(x[i] * scale * w[i]);
}

static __device__ __forceinline__ float rope_val(const float* __restrict__ src,
                                                 int d, int s) {
  int j  = d - NOPE_;
  int jm = j & 31;
  float inv = __expf(-9.210340371976184f * (float)jm * (1.0f / 32.0f)); // 10000^(-jm/32)
  float ang = (float)s * inv;
  float c = __cosf(ang), sn = __sinf(ang);
  float x  = src[d];
  float rh = (j < 32) ? -src[d + 32] : src[d - 32];
  return x * c + rh * sn;
}

// q_raw (B,S,H,192) f32 -> q_full (B,H,S,192) bf16 with RoPE on last 64 dims
__global__ void rope_pack_q(const float* __restrict__ q_raw,
                            bf16_t* __restrict__ qf) {
  size_t idx = (size_t)blockIdx.x * blockDim.x + threadIdx.x;
  if (idx >= (size_t)BS_ * QOUT_) return;
  int d = (int)(idx % QD_);
  size_t rest = idx / QD_;
  int h = (int)(rest % H_);
  size_t bs = rest / H_;
  int s = (int)(bs % S_);
  int b = (int)(bs / S_);
  const float* src = q_raw + (bs * H_ + h) * QD_;
  float val = (d < NOPE_) ? src[d] : rope_val(src, d, s);
  qf[(((size_t)b * H_ + h) * S_ + s) * QD_ + d] = (bf16_t)val;
}

// kv_raw (B,S,KVH,320) f32 -> k_full (B,KVH,S,192) bf16 (RoPE on 128..191)
//                             vT     (B,KVH,128,S) bf16 (transposed V)
__global__ void rope_pack_kv(const float* __restrict__ kv_raw,
                             bf16_t* __restrict__ kf, bf16_t* __restrict__ vt) {
  size_t idx = (size_t)blockIdx.x * blockDim.x + threadIdx.x;
  if (idx >= (size_t)BS_ * KVOUT_) return;
  int d = (int)(idx % KVD_);
  size_t rest = idx / KVD_;
  int g = (int)(rest % KVH_);
  size_t bs = rest / KVH_;
  int s = (int)(bs % S_);
  int b = (int)(bs / S_);
  const float* src = kv_raw + (bs * KVH_ + g) * KVD_;
  if (d < QD_) {
    float val = (d < NOPE_) ? src[d] : rope_val(src, d, s);
    kf[(((size_t)b * KVH_ + g) * S_ + s) * QD_ + d] = (bf16_t)val;
  } else {
    int dv = d - QD_;
    vt[(((size_t)b * KVH_ + g) * VD_ + dv) * (size_t)S_ + s] = (bf16_t)src[d];
  }
}

// ---------------------------------------------------------------------------
// Flash attention: 4 waves/block share one (b,h); each wave owns a 16-query
// tile. K (32x192) and V^T (128x32) tiles are TDM-staged into LDS per 32-key
// block; online softmax; PV via WMMA. ctx written as (B,S,H*VD) bf16.
// ---------------------------------------------------------------------------
__global__ __launch_bounds__(128) void mla_attention(
    const bf16_t* __restrict__ qf,   // (B,H,S,192)
    const bf16_t* __restrict__ kf,   // (B,KVH,S,192)
    const bf16_t* __restrict__ vt,   // (B,KVH,128,S)
    const float*  __restrict__ mask, // (S,S)
    bf16_t* __restrict__ ctx)        // (B,S,H*128)
{
#if HAVE_TDM
  __shared__ __align__(16) bf16_t Ks[32 * QD_];    // 12 KB
  __shared__ __align__(16) bf16_t Vs[VD_ * 32];    //  8 KB
#endif
  __shared__ __align__(16) bf16_t pshm[4][16 * 32];
  const int wave = threadIdx.x >> 5;
  const int lane = threadIdx.x & 31;
  const int gw = blockIdx.x * 4 + wave;
  const int qt = gw & (S_ / 16 - 1);
  const int h  = (gw >> 7) & (H_ - 1);
  const int b  = gw >> 11;
  const int q0 = qt * 16;
  const int kvh = h >> 2;

  const bf16_t* qbase = qf + (((size_t)b * H_ + h) * S_ + q0) * QD_;
  const bf16_t* kbase = kf + (((size_t)b * KVH_ + kvh) * S_) * QD_;
  const bf16_t* vbase = vt + (((size_t)b * KVH_ + kvh) * VD_) * (size_t)S_;

  v16bf qfrag[6];
  for (int d = 0; d < 6; ++d) qfrag[d] = load_a_frag(qbase, QD_, 0, d * 32);

  v8f o[8] = {};
  float mrow[8], lrow[8];
  for (int i = 0; i < 8; ++i) { mrow[i] = -1e30f; lrow[i] = 0.f; }

  const int col = lane & 15;
  const int rb  = (lane >> 4) << 3;
  const float scale = 0.07216878364870322f;   // 1/sqrt(192)
  bf16_t* pw = pshm[wave];
#if HAVE_TDM
  const unsigned ldsK = (unsigned)(unsigned long long)(uintptr_t)Ks;
  const unsigned ldsV = (unsigned)(unsigned long long)(uintptr_t)Vs;
#endif

  for (int kb = 0; kb < S_; kb += 32) {
#if HAVE_TDM
    if (wave == 0) {
      // K tile: 32 rows x 192 dims (row-major, stride 192)
      tdm_load_2d(ldsK,
                  (unsigned long long)(uintptr_t)(kbase + (size_t)kb * QD_),
                  (unsigned)QD_, (unsigned)(S_ - kb),
                  (unsigned long long)QD_, (unsigned)QD_, 32u);
      // V^T tile: 128 rows (dims) x 32 keys (row-major, stride S)
      tdm_load_2d(ldsV,
                  (unsigned long long)(uintptr_t)(vbase + kb),
                  (unsigned)(S_ - kb), (unsigned)VD_,
                  (unsigned long long)S_, 32u, (unsigned)VD_);
      __builtin_amdgcn_s_wait_tensorcnt(0);
    }
    __syncthreads();
#endif
    v8f s0 = {}, s1 = {};
    for (int d = 0; d < 6; ++d) {
#if HAVE_TDM
      v16bf k0f = load_b_frag(Ks, QD_, 0,  d * 32);
      v16bf k1f = load_b_frag(Ks, QD_, 16, d * 32);
#else
      v16bf k0f = load_b_frag(kbase, QD_, kb,      d * 32);
      v16bf k1f = load_b_frag(kbase, QD_, kb + 16, d * 32);
#endif
      s0 = wmma_bf16(qfrag[d], k0f, s0);
      s1 = wmma_bf16(qfrag[d], k1f, s1);
    }
    float p0[8], p1[8], alpha[8];
    for (int i = 0; i < 8; ++i) {
      const float* mp = mask + (size_t)(q0 + rb + i) * S_ + kb;
      p0[i] = s0[i] * scale + mp[col];
      p1[i] = s1[i] * scale + mp[16 + col];
    }
    for (int i = 0; i < 8; ++i) {
      float rmax = fmaxf(p0[i], p1[i]);
      for (int mm = 1; mm < 16; mm <<= 1)
        rmax = fmaxf(rmax, __shfl_xor(rmax, mm, 32));
      float nm = fmaxf(mrow[i], rmax);
      alpha[i] = __expf(mrow[i] - nm);
      mrow[i] = nm;
      p0[i] = __expf(p0[i] - nm);
      p1[i] = __expf(p1[i] - nm);
      float rs = p0[i] + p1[i];
      for (int mm = 1; mm < 16; mm <<= 1) rs += __shfl_xor(rs, mm, 32);
      lrow[i] = lrow[i] * alpha[i] + rs;
    }
    for (int t = 0; t < 8; ++t)
      for (int i = 0; i < 8; ++i) o[t][i] *= alpha[i];
    for (int i = 0; i < 8; ++i) {
      pw[(rb + i) * 32 + col]      = (bf16_t)p0[i];
      pw[(rb + i) * 32 + 16 + col] = (bf16_t)p1[i];
    }
    v16bf pa = load_a_frag(pw, 32, 0, 0);
    for (int t = 0; t < 8; ++t) {
#if HAVE_TDM
      v16bf vfrag = load_b_frag(Vs, 32, t * 16, 0);
#else
      v16bf vfrag = load_b_frag(vbase, S_, t * 16, kb);
#endif
      o[t] = wmma_bf16(pa, vfrag, o[t]);
    }
#if HAVE_TDM
    __syncthreads();   // protect Ks/Vs before next TDM overwrite
#endif
  }
  for (int t = 0; t < 8; ++t)
    for (int i = 0; i < 8; ++i) {
      float v = o[t][i] / lrow[i];
      ctx[(((size_t)b * S_ + q0 + rb + i) * H_ + h) * VD_ + t * 16 + col] =
          (bf16_t)v;
    }
}

// ---------------------------------------------------------------------------
// Host-side orchestration
// ---------------------------------------------------------------------------
extern "C" void kernel_launch(void* const* d_in, const int* in_sizes, int n_in,
                              void* d_out, int out_size, void* d_ws, size_t ws_size,
                              hipStream_t stream) {
  const float* x    = (const float*)d_in[0];
  const float* mask = (const float*)d_in[1];
  const float* qan  = (const float*)d_in[2];
  const float* kvan = (const float*)d_in[3];
  const float* Wqa  = (const float*)d_in[4];
  const float* Wkva = (const float*)d_in[5];
  const float* Wqb  = (const float*)d_in[6];
  const float* Wkvb = (const float*)d_in[7];
  const float* Wo   = (const float*)d_in[8];
  float* out = (float*)d_out;

  char* wsp = (char*)d_ws;
  auto alloc = [&](size_t bytes) -> char* {
    char* p = wsp;
    wsp += (bytes + 255) & ~(size_t)255;
    return p;
  };
  bf16_t* x_bf    = (bf16_t*)alloc((size_t)BS_ * HID_ * 2);
  bf16_t* WqaT    = (bf16_t*)alloc((size_t)QR_ * HID_ * 2);
  bf16_t* WkvaT   = (bf16_t*)alloc((size_t)KVR_ * HID_ * 2);
  bf16_t* WqbT    = (bf16_t*)alloc((size_t)QOUT_ * QR_ * 2);
  bf16_t* WkvbT   = (bf16_t*)alloc((size_t)KVOUT_ * KVR_ * 2);
  bf16_t* WoT     = (bf16_t*)alloc((size_t)HID_ * HID_ * 2);
  float*  qlat_r  = (float*)alloc((size_t)BS_ * QR_ * 4);
  float*  kvlat_r = (float*)alloc((size_t)BS_ * KVR_ * 4);
  bf16_t* qlat_bf = (bf16_t*)alloc((size_t)BS_ * QR_ * 2);
  bf16_t* kvlat_bf= (bf16_t*)alloc((size_t)BS_ * KVR_ * 2);
  float*  q_raw   = (float*)alloc((size_t)BS_ * QOUT_ * 4);
  float*  kv_raw  = (float*)alloc((size_t)BS_ * KVOUT_ * 4);
  bf16_t* q_full  = (bf16_t*)alloc((size_t)B_ * H_ * S_ * QD_ * 2);
  bf16_t* k_full  = (bf16_t*)alloc((size_t)B_ * KVH_ * S_ * QD_ * 2);
  bf16_t* vT      = (bf16_t*)alloc((size_t)B_ * KVH_ * VD_ * S_ * 2);
  bf16_t* ctx_bf  = (bf16_t*)alloc((size_t)BS_ * HID_ * 2);

  auto nb = [](size_t n) { return (unsigned)((n + 255) / 256); };

  // 1) bf16 conversions / weight transposes
  f32_to_bf16<<<nb((size_t)BS_ * HID_), 256, 0, stream>>>(x, x_bf, (size_t)BS_ * HID_);
  transpose_to_bf16<<<nb((size_t)HID_ * QR_),    256, 0, stream>>>(Wqa,  WqaT,  HID_, QR_);
  transpose_to_bf16<<<nb((size_t)HID_ * KVR_),   256, 0, stream>>>(Wkva, WkvaT, HID_, KVR_);
  transpose_to_bf16<<<nb((size_t)QR_ * QOUT_),   256, 0, stream>>>(Wqb,  WqbT,  QR_, QOUT_);
  transpose_to_bf16<<<nb((size_t)KVR_ * KVOUT_), 256, 0, stream>>>(Wkvb, WkvbT, KVR_, KVOUT_);
  transpose_to_bf16<<<nb((size_t)HID_ * HID_),   256, 0, stream>>>(Wo,   WoT,   HID_, HID_);

  // 2) low-rank projections
  GEMM_KERNEL<<<dim3(QR_ / 64, BS_ / 64), 128, 0, stream>>>(x_bf, WqaT, qlat_r, BS_, QR_, HID_);
  GEMM_KERNEL<<<dim3(KVR_ / 64, BS_ / 64), 128, 0, stream>>>(x_bf, WkvaT, kvlat_r, BS_, KVR_, HID_);

  // 3) RMS norms
  rmsnorm_bf16<<<BS_, 256, 0, stream>>>(qlat_r,  qan,  qlat_bf,  QR_);
  rmsnorm_bf16<<<BS_, 256, 0, stream>>>(kvlat_r, kvan, kvlat_bf, KVR_);

  // 4) up-projections
  GEMM_KERNEL<<<dim3(QOUT_ / 64, BS_ / 64), 128, 0, stream>>>(qlat_bf, WqbT, q_raw, BS_, QOUT_, QR_);
  GEMM_KERNEL<<<dim3(KVOUT_ / 64, BS_ / 64), 128, 0, stream>>>(kvlat_bf, WkvbT, kv_raw, BS_, KVOUT_, KVR_);

  // 5) RoPE + attention layouts (q_full / k_full / vT)
  rope_pack_q <<<nb((size_t)BS_ * QOUT_),  256, 0, stream>>>(q_raw,  q_full);
  rope_pack_kv<<<nb((size_t)BS_ * KVOUT_), 256, 0, stream>>>(kv_raw, k_full, vT);

  // 6) flash attention: B*H*(S/16) = 4096 waves -> 1024 blocks of 4 waves
  mla_attention<<<(B_ * H_ * (S_ / 16)) / 4, 128, 0, stream>>>(q_full, k_full, vT, mask, ctx_bf);

  // 7) output projection -> f32 d_out
  GEMM_KERNEL<<<dim3(HID_ / 64, BS_ / 64), 128, 0, stream>>>(ctx_bf, WoT, out, BS_, HID_, HID_);
}
